// MicroEncoder_62586263437422
// MI455X (gfx1250) — compile-verified, matrix operand order checked by reading, not run on hardware
//
#include <hip/hip_runtime.h>
#include <cstdint>
#include <cstddef>

#define B_SZ     8
#define SEQ      2048
#define NTOK     (B_SZ * SEQ)      // 16384
#define DM       256               // D_MODEL
#define DI       512               // D_INNER
#define DS       64                // D_STATE
#define HD       64                // HEADDIM
#define NH       8                 // NHEADS
#define DCONV    4
#define CDIM     640               // CONV_DIM = DI + 2*DS
#define DPROJ    1160              // D_IN_PROJ
#define NPAD     1184              // DPROJ padded to /32 (37 tiles of 32)

typedef __bf16 bf16_t;
typedef bf16_t v16bf __attribute__((ext_vector_type(16)));
typedef bf16_t v8bf  __attribute__((ext_vector_type(8)));
typedef float  v8f   __attribute__((ext_vector_type(8)));

// ---------------------------------------------------------------- WMMA GEMM
// C(M x N) = A(M x K, row-major bf16) * B(N x K, row-major bf16)^T, f32 accum.
// One wave computes a 32x32 tile (2x2 WMMA register blocking, 4 WMMA / k-step).
// Fragment layout per CDNA5 ISA 7.12.2: lane l -> row/col l%16;
// half0 lanes hold K 0-7 & 16-23, half1 lanes hold K 8-15 & 24-31.
__device__ __forceinline__ v16bf ld_frag(const bf16_t* p0) {
  v8bf lo = *(const v8bf*)(p0);
  v8bf hi = *(const v8bf*)(p0 + 16);
  return __builtin_shufflevector(lo, hi, 0,1,2,3,4,5,6,7,8,9,10,11,12,13,14,15);
}

__device__ __forceinline__ v8f wmma_bf16(v16bf a, v16bf b, v8f c) {
  return __builtin_amdgcn_wmma_f32_16x16x32_bf16(false, a, false, b, (short)0, c,
                                                 false, false);
}

template <int K>
__global__ void gemm_bf16_wmma(const bf16_t* __restrict__ A,
                               const bf16_t* __restrict__ Bw,
                               float* __restrict__ C,
                               int lda, int ldb, int ldc) {
  const int n0   = blockIdx.x * 32;
  const int m0   = blockIdx.y * 32;
  const int lane = threadIdx.x & 31;
  const int half = lane >> 4;
  const int lm   = lane & 15;
  const bf16_t* a0 = A  + (size_t)(m0 + lm) * lda;
  const bf16_t* a1 = A  + (size_t)(m0 + 16 + lm) * lda;
  const bf16_t* b0 = Bw + (size_t)(n0 + lm) * ldb;
  const bf16_t* b1 = Bw + (size_t)(n0 + 16 + lm) * ldb;
  v8f acc00 = {}, acc01 = {}, acc10 = {}, acc11 = {};
#pragma unroll
  for (int kk = 0; kk < K; kk += 32) {
    const int ka = kk + half * 8;
    v16bf fa0 = ld_frag(a0 + ka);
    v16bf fa1 = ld_frag(a1 + ka);
    v16bf fb0 = ld_frag(b0 + ka);
    v16bf fb1 = ld_frag(b1 + ka);
    acc00 = wmma_bf16(fa0, fb0, acc00);
    acc01 = wmma_bf16(fa0, fb1, acc01);
    acc10 = wmma_bf16(fa1, fb0, acc10);
    acc11 = wmma_bf16(fa1, fb1, acc11);
  }
  const int rbase = m0 + half * 8;
  float* c0 = C + (size_t)rbase * ldc + n0 + lm;
  float* c1 = C + (size_t)(rbase + 16) * ldc + n0 + lm;
#pragma unroll
  for (int r = 0; r < 8; ++r) {
    c0[(size_t)r * ldc]      = acc00[r];
    c0[(size_t)r * ldc + 16] = acc01[r];
    c1[(size_t)r * ldc]      = acc10[r];
    c1[(size_t)r * ldc + 16] = acc11[r];
  }
}

// ---------------------------------------------------------------- helpers
__global__ void f32_to_bf16_k(const float* __restrict__ in, bf16_t* __restrict__ out, int n) {
  const int i = blockIdx.x * blockDim.x + threadIdx.x;
  if (i < n) out[i] = (bf16_t)in[i];
}

__global__ void pad_win_bf16_k(const float* __restrict__ Win, bf16_t* __restrict__ out) {
  const int i = blockIdx.x * blockDim.x + threadIdx.x;   // NPAD * DM
  if (i >= NPAD * DM) return;
  const int n = i / DM;
  out[i] = (n < DPROJ) ? (bf16_t)Win[i] : (bf16_t)0.f;
}

__global__ void combine_bf16_k(const float* __restrict__ a, const float* __restrict__ b,
                               bf16_t* __restrict__ out, int n) {
  const int i = blockIdx.x * blockDim.x + threadIdx.x;
  if (i < n) out[i] = (bf16_t)(a[i] + b[i]);
}

// causal depthwise conv (4 taps) + SiLU, direction-aware
__global__ void conv_silu_k(const float* __restrict__ Yin, const float* __restrict__ Wc,
                            const float* __restrict__ bc, float* __restrict__ xBC, int dir) {
  const int i = blockIdx.x * blockDim.x + threadIdx.x;   // NTOK * CDIM
  if (i >= NTOK * CDIM) return;
  const int c   = i % CDIM;
  const int tok = i / CDIM;
  const int t   = tok % SEQ;
  const int b   = tok / SEQ;
  float acc = bc[c];
#pragma unroll
  for (int k = 0; k < DCONV; ++k) {
    const int tk = dir ? (t + (DCONV - 1) - k) : (t - (DCONV - 1) + k);
    if (tk >= 0 && tk < SEQ)
      acc += Wc[c * DCONV + k] * Yin[((size_t)b * SEQ + tk) * NPAD + DI + c];
  }
  xBC[i] = acc / (1.f + expf(-acc));   // silu
}

__global__ void dt_dA_k(const float* __restrict__ Yin, const float* __restrict__ dt_bias,
                        const float* __restrict__ A_log, float* __restrict__ dt,
                        float* __restrict__ dA) {
  const int i = blockIdx.x * blockDim.x + threadIdx.x;   // NTOK * NH
  if (i >= NTOK * NH) return;
  const int hh  = i % NH;
  const int tok = i / NH;
  const float raw = Yin[(size_t)tok * NPAD + DI + CDIM + hh] + dt_bias[hh];
  const float dtv = raw > 20.f ? raw : log1pf(expf(raw));  // softplus
  dt[i] = dtv;
  dA[i] = expf(-expf(A_log[hh]) * dtv);
}

// selective scan: one block per (batch, head); 64 threads, thread p owns h[p][0..63].
// 4 independent y-accumulators shorten the dependent FMA chain; prefetch pulls the
// xBC row 16 steps ahead into cache (gfx1250 global_prefetch path).
__global__ void scan_k(const float* __restrict__ xBC, const float* __restrict__ dt,
                       const float* __restrict__ dA, const float* __restrict__ Dp,
                       float* __restrict__ ybuf, int dir) {
  const int bh = blockIdx.x;
  const int b  = bh / NH, hh = bh % NH;
  const int p  = threadIdx.x;                 // 0..63
  __shared__ float sB[DS], sC[DS];
  float h[DS];
#pragma unroll
  for (int n = 0; n < DS; ++n) h[n] = 0.f;
  const float Dh = Dp[hh];
  for (int step = 0; step < SEQ; ++step) {
    const int t = dir ? (SEQ - 1 - step) : step;
    const size_t tok = (size_t)b * SEQ + t;
    const float* row = xBC + tok * CDIM;
    // prefetch 16 steps ahead
    {
      const int sp = step + 16 < SEQ ? step + 16 : SEQ - 1;
      const int tp = dir ? (SEQ - 1 - sp) : sp;
      __builtin_prefetch(xBC + ((size_t)b * SEQ + tp) * CDIM + p * 8, 0, 1);
    }
    sB[p] = row[DI + p];
    sC[p] = row[DI + DS + p];
    __syncthreads();
    const float xv   = row[hh * HD + p];
    const float dtv  = dt[tok * NH + hh];
    const float dAv  = dA[tok * NH + hh];
    const float coef = dtv * xv;
    float y0 = 0.f, y1 = 0.f, y2 = 0.f, y3 = 0.f;
#pragma unroll
    for (int n = 0; n < DS; n += 4) {
      h[n + 0] = dAv * h[n + 0] + coef * sB[n + 0]; y0 += h[n + 0] * sC[n + 0];
      h[n + 1] = dAv * h[n + 1] + coef * sB[n + 1]; y1 += h[n + 1] * sC[n + 1];
      h[n + 2] = dAv * h[n + 2] + coef * sB[n + 2]; y2 += h[n + 2] * sC[n + 2];
      h[n + 3] = dAv * h[n + 3] + coef * sB[n + 3]; y3 += h[n + 3] * sC[n + 3];
    }
    ybuf[tok * DI + hh * HD + p] = ((y0 + y1) + (y2 + y3)) + Dh * xv;
    __syncthreads();
  }
}

// y * silu(z), RMSNorm over 512 channels, emit bf16 for out-projection
__global__ void gate_norm_k(const float* __restrict__ ybuf, const float* __restrict__ Yin,
                            const float* __restrict__ norm_w, bf16_t* __restrict__ normbf) {
  const int tok = blockIdx.x;
  const int tid = threadIdx.x;                // 128
  __shared__ float red[128];
  float v[4];
  float ss = 0.f;
#pragma unroll
  for (int j = 0; j < 4; ++j) {
    const int c   = tid + 128 * j;
    const float y = ybuf[(size_t)tok * DI + c];
    const float z = Yin[(size_t)tok * NPAD + c];
    const float g = z / (1.f + expf(-z));
    v[j] = y * g;
    ss  += v[j] * v[j];
  }
  red[tid] = ss;
  __syncthreads();
  for (int s = 64; s > 0; s >>= 1) {
    if (tid < s) red[tid] += red[tid + s];
    __syncthreads();
  }
  const float scale = rsqrtf(red[0] / (float)DI + 1e-5f);
#pragma unroll
  for (int j = 0; j < 4; ++j) {
    const int c = tid + 128 * j;
    normbf[(size_t)tok * DI + c] = (bf16_t)(v[j] * scale * norm_w[c]);
  }
}

__global__ void hmean_k(const float* __restrict__ hf, const float* __restrict__ hb,
                        float* __restrict__ H) {
  const int b   = blockIdx.x / DM;
  const int d   = blockIdx.x % DM;
  const int tid = threadIdx.x;                // 256
  __shared__ float red[256];
  float s = 0.f;
#pragma unroll
  for (int j = 0; j < SEQ / 256; ++j) {
    const int t = tid + 256 * j;
    const size_t idx = ((size_t)b * SEQ + t) * DM + d;
    s += hf[idx] + hb[idx];
  }
  red[tid] = s;
  __syncthreads();
  for (int st = 128; st > 0; st >>= 1) {
    if (tid < st) red[tid] += red[tid + st];
    __syncthreads();
  }
  if (tid == 0) H[(size_t)b * DM + d] = red[0] / (float)SEQ;
}

// ---------------------------------------------------------------- launcher
extern "C" void kernel_launch(void* const* d_in, const int* in_sizes, int n_in,
                              void* d_out, int out_size, void* d_ws, size_t ws_size,
                              hipStream_t stream) {
  (void)in_sizes; (void)n_in; (void)out_size; (void)ws_size;

  const float* x_in = (const float*)d_in[0];

  // carve workspace (256B aligned)
  char* p = (char*)d_ws;
  auto carve = [&](size_t bytes) -> void* {
    void* r = (void*)p;
    p += (bytes + 255) & ~(size_t)255;
    return r;
  };
  bf16_t* xbf    = (bf16_t*)carve((size_t)NTOK * DM * 2);
  bf16_t* wInBf  = (bf16_t*)carve((size_t)NPAD * DM * 2);
  bf16_t* wOutBf = (bf16_t*)carve((size_t)DM * DI * 2);
  float*  Yin    = (float*) carve((size_t)NTOK * NPAD * 4);
  float*  xBCact = (float*) carve((size_t)NTOK * CDIM * 4);
  float*  dtB    = (float*) carve((size_t)NTOK * NH * 4);
  float*  dAB    = (float*) carve((size_t)NTOK * NH * 4);
  float*  ybuf   = (float*) carve((size_t)NTOK * DI * 4);
  bf16_t* normbf = (bf16_t*)carve((size_t)NTOK * DI * 2);
  float*  outF   = (float*) carve((size_t)NTOK * DM * 4);
  float*  outB   = (float*) carve((size_t)NTOK * DM * 4);

  float* out_hf = (float*)d_out;
  float* out_hb = out_hf + (size_t)NTOK * DM;
  float* out_H  = out_hb + (size_t)NTOK * DM;

  const int T256 = 256;
  auto blocks = [](long n, int t) { return (int)((n + t - 1) / t); };

  // layer-0 input -> bf16
  f32_to_bf16_k<<<blocks((long)NTOK * DM, T256), T256, 0, stream>>>(x_in, xbf, NTOK * DM);

  for (int layer = 0; layer < 4; ++layer) {
    for (int dir = 0; dir < 2; ++dir) {
      const int base = 1 + dir * 8;  // fwd params at d_in[1..8], bwd at d_in[9..16]
      const float* Win     = (const float*)d_in[base + 0] + (size_t)layer * DPROJ * DM;
      const float* Wconv   = (const float*)d_in[base + 1] + (size_t)layer * CDIM * DCONV;
      const float* bconv   = (const float*)d_in[base + 2] + (size_t)layer * CDIM;
      const float* dt_bias = (const float*)d_in[base + 3] + (size_t)layer * NH;
      const float* A_log   = (const float*)d_in[base + 4] + (size_t)layer * NH;
      const float* Dp      = (const float*)d_in[base + 5] + (size_t)layer * NH;
      const float* norm_w  = (const float*)d_in[base + 6] + (size_t)layer * DI;
      const float* Wout    = (const float*)d_in[base + 7] + (size_t)layer * DM * DI;

      // weights -> bf16 (Win zero-padded to 1184 rows)
      pad_win_bf16_k<<<blocks((long)NPAD * DM, T256), T256, 0, stream>>>(Win, wInBf);
      f32_to_bf16_k<<<blocks((long)DM * DI, T256), T256, 0, stream>>>(Wout, wOutBf, DM * DI);

      // in-projection: (16384 x 256) @ (1184 x 256)^T -> (16384 x 1184)
      dim3 g1(NPAD / 32, NTOK / 32);
      gemm_bf16_wmma<DM><<<g1, 32, 0, stream>>>(xbf, wInBf, Yin, DM, DM, NPAD);

      // conv + silu ; dt/dA
      conv_silu_k<<<blocks((long)NTOK * CDIM, T256), T256, 0, stream>>>(Yin, Wconv, bconv,
                                                                        xBCact, dir);
      dt_dA_k<<<blocks((long)NTOK * NH, T256), T256, 0, stream>>>(Yin, dt_bias, A_log,
                                                                  dtB, dAB);

      // selective scan (64 blocks of 64 threads)
      scan_k<<<B_SZ * NH, 64, 0, stream>>>(xBCact, dtB, dAB, Dp, ybuf, dir);

      // gate + RMSNorm -> bf16
      gate_norm_k<<<NTOK, 128, 0, stream>>>(ybuf, Yin, norm_w, normbf);

      // out-projection: (16384 x 512) @ (256 x 512)^T -> (16384 x 256)
      float* cout = (layer < 3) ? (dir ? outB : outF) : (dir ? out_hb : out_hf);
      dim3 g2(DM / 32, NTOK / 32);
      gemm_bf16_wmma<DI><<<g2, 32, 0, stream>>>(normbf, wOutBf, cout, DI, DI, DM);
    }
    if (layer < 3)
      combine_bf16_k<<<blocks((long)NTOK * DM, T256), T256, 0, stream>>>(outF, outB, xbf,
                                                                         NTOK * DM);
  }

  // H = mean over L of (h_fwd + h_bwd)
  hmean_k<<<B_SZ * DM, 256, 0, stream>>>(out_hf, out_hb, out_H);
}
